// EquivariantGNN_21990232555992
// MI455X (gfx1250) — compile-verified, matrix operand order checked by reading, not run on hardware
//
#include <hip/hip_runtime.h>
#include <cstdint>
#include <cstddef>

// ---------------------------------------------------------------------------
// MI455X (gfx1250) implementation of the equivariant GNN forward pass.
//
// Roofline reasoning:
//  * Edge MLP (8->64->64->{64,128 used}) over 640k edges x 3 blocks ~ 44 GFLOP
//    of shared-weight GEMM -> v_wmma_f32_16x16x32_f16 (f16 in, f32 accum).
//  * Scatter (segment_sum) accumulators are 20000 nodes x <=256 f32 ~ 20 MB,
//    L2-resident (192 MB L2) -> global f32 atomics are cheap; fusing the MLP
//    with the scatter keeps the 0.5 GB/block "w" tensor in LDS and saves
//    ~3 GB of HBM traffic per call.
//  * Node-side GEMMs are ~0.3 GFLOP total -> plain VALU kernels.
//  * silu uses v_exp_f32 + v_rcp_f32 (fast transcendental path) instead of the
//    IEEE division expansion: activations are truncated to f16 anyway.
// ---------------------------------------------------------------------------

typedef __attribute__((ext_vector_type(16))) _Float16 v16h;
typedef __attribute__((ext_vector_type(8)))  float    v8f;

#define R_CUT_F 5.0f
#define MULC    32
#define NBASIS  8

__device__ __forceinline__ float siluf(float x) {
  // x / (1 + exp(-x)) with hardware v_rcp_f32 (1 ulp) -- avoids the precise
  // div_scale/div_fmas expansion that otherwise dominates the epilogue.
  return x * __builtin_amdgcn_rcpf(1.0f + __expf(-x));
}

// Build a 16x32 f16 WMMA fragment from a row-major [16][stride] LDS tile.
// Per the CDNA5 ISA 16-bit A layout: lanes 0-15 hold K={k0..k0+7, k0+16..k0+23},
// lanes 16-31 hold K={k0+8..k0+15, k0+24..k0+31} for row (lane&15).
// B fragments (weights stored transposed [N][K]) use the mirrored layout with
// lane = column N, same K striping.
__device__ __forceinline__ v16h frag_ld(const _Float16* __restrict__ base,
                                        int stride, int k0, int lane) {
  const int r  = lane & 15;
  const int hh = (lane >> 4) << 3;     // 0 or 8
  const _Float16* p = base + r * stride + k0 + hh;
  v16h f;
#pragma unroll
  for (int i = 0; i < 8; ++i) f[i] = p[i];
#pragma unroll
  for (int i = 0; i < 8; ++i) f[8 + i] = p[16 + i];
  return f;
}

__device__ __forceinline__ v8f wmma_f16(v16h a, v16h b, v8f c) {
  return __builtin_amdgcn_wmma_f32_16x16x32_f16(
      /*neg_a=*/false, a, /*neg_b=*/false, b,
      /*c_mod=*/(short)0, c, /*reuse_a=*/false, /*reuse_b=*/false);
}

// ---------------------------------------------------------------------------
// Utility: zero fill
// ---------------------------------------------------------------------------
__global__ void fill_zero_kernel(float* __restrict__ p, int n) {
  int i = blockIdx.x * 256 + threadIdx.x;
  if (i < n) p[i] = 0.0f;
}

// ---------------------------------------------------------------------------
// Edge geometry: Y1 (f32 [E][3]) and radial embedding (f16 [E][8])
// ---------------------------------------------------------------------------
__global__ __launch_bounds__(256)
void edge_geom_kernel(const float* __restrict__ pos,
                      const int* __restrict__ ei,
                      const int* __restrict__ ej,
                      const float* __restrict__ bpi,
                      float* __restrict__ Y1,
                      _Float16* __restrict__ embh,
                      int nE) {
  int e = blockIdx.x * 256 + threadIdx.x;
  if (e >= nE) return;
  int i = ei[e], j = ej[e];
  float vx = pos[j * 3 + 0] - pos[i * 3 + 0];
  float vy = pos[j * 3 + 1] - pos[i * 3 + 1];
  float vz = pos[j * 3 + 2] - pos[i * 3 + 2];
  float r  = sqrtf(vx * vx + vy * vy + vz * vz);
  float inv = __builtin_amdgcn_rcpf(fmaxf(r, 1e-12f));
  const float Y1C = 0.48860251190291992f;  // sqrt(3/(4*pi))
  Y1[e * 3 + 0] = Y1C * vx * inv;
  Y1[e * 3 + 1] = Y1C * vy * inv;
  Y1[e * 3 + 2] = Y1C * vz * inv;
  // polynomial cutoff envelope, P = 6
  float t  = r / R_CUT_F;
  float t2 = t * t, t3 = t2 * t, t6 = t3 * t3;
  float env = 1.0f - 28.0f * t6 + 48.0f * t6 * t - 21.0f * t6 * t2;
  env = (t < 1.0f) ? env : 0.0f;
  float pre = (2.0f / R_CUT_F) * env * inv;
#pragma unroll
  for (int b = 0; b < NBASIS; ++b) {
    float v = (r > 0.0f) ? pre * __sinf(bpi[b] * r / R_CUT_F) : 0.0f;
    embh[e * NBASIS + b] = (_Float16)v;
  }
}

// ---------------------------------------------------------------------------
// Node embedding: x0[n][u] = emb_table[type[n]][u]
// ---------------------------------------------------------------------------
__global__ __launch_bounds__(256)
void node_embed_kernel(const float* __restrict__ embT,
                       const int* __restrict__ types,
                       float* __restrict__ x0, int n32) {
  int i = blockIdx.x * 256 + threadIdx.x;
  if (i >= n32) return;
  int node = i >> 5, u = i & 31;
  x0[i] = embT[types[node] * MULC + u];
}

// ---------------------------------------------------------------------------
// Per-node pre-compute: h0 = x0 @ W1_0, h1 = einsum(x1, W1_1),
// sc0[n][64] = x0 . Wsc0[:, t, :], sc1[n][32][3] = einsum(x1, Wsc1[:, t, :])
// ---------------------------------------------------------------------------
template <bool HAS_X1>
__global__ __launch_bounds__(64)
void node_pre_kernel(const float* __restrict__ x0,
                     const float* __restrict__ x1,
                     const int* __restrict__ types,
                     const float* __restrict__ W1_0,
                     const float* __restrict__ W1_1,
                     const float* __restrict__ Wsc0,
                     const float* __restrict__ Wsc1,
                     float* __restrict__ h0,
                     float* __restrict__ h1,
                     float* __restrict__ sc0,
                     float* __restrict__ sc1,
                     int n) {
  int node = blockIdx.x;
  if (node >= n) return;
  int tid = threadIdx.x;
  __shared__ float x0s[32];
  __shared__ float x1s[96];
  if (tid < 32) x0s[tid] = x0[node * 32 + tid];
  if (HAS_X1) {
    x1s[tid] = x1[node * 96 + tid];
    if (tid < 32) x1s[64 + tid] = x1[node * 96 + 64 + tid];
  }
  __syncthreads();
  int t = types[node];
  float acc = 0.0f;
#pragma unroll
  for (int u = 0; u < 32; ++u) acc += x0s[u] * Wsc0[(u * 4 + t) * 64 + tid];
  sc0[node * 64 + tid] = acc;
  if (tid < 32) {
    float a = 0.0f;
#pragma unroll
    for (int u = 0; u < 32; ++u) a += x0s[u] * W1_0[u * 32 + tid];
    h0[node * 32 + tid] = a;
  }
  if (HAS_X1) {
    if (tid < 32) {
      float a0v = 0.0f, a1v = 0.0f, a2v = 0.0f;
#pragma unroll
      for (int u = 0; u < 32; ++u) {
        float wv = W1_1[u * 32 + tid];
        a0v += x1s[u * 3 + 0] * wv;
        a1v += x1s[u * 3 + 1] * wv;
        a2v += x1s[u * 3 + 2] * wv;
      }
      h1[(node * 32 + tid) * 3 + 0] = a0v;
      h1[(node * 32 + tid) * 3 + 1] = a1v;
      h1[(node * 32 + tid) * 3 + 2] = a2v;
    } else {
      int w = tid - 32;
      float a0v = 0.0f, a1v = 0.0f, a2v = 0.0f;
#pragma unroll
      for (int u = 0; u < 32; ++u) {
        float wv = Wsc1[(u * 4 + t) * 32 + w];
        a0v += x1s[u * 3 + 0] * wv;
        a1v += x1s[u * 3 + 1] * wv;
        a2v += x1s[u * 3 + 2] * wv;
      }
      sc1[(node * 32 + w) * 3 + 0] = a0v;
      sc1[(node * 32 + w) * 3 + 1] = a1v;
      sc1[(node * 32 + w) * 3 + 2] = a2v;
    }
  }
}

// ---------------------------------------------------------------------------
// Fused per-edge kernel: WMMA MLP (8->64->64->NOUT) + message + atomic scatter.
// 256 threads = 8 waves; each wave owns a 16-edge M-tile; WG tile = 128 edges.
//
// LDS layout (halves), 52 KB total:
//   phase 1: emb[0,4096) W1T[4096,6144) W2T[6144,10240)
//            hstg[10240,18432) W3T[18432,26624)
//   phase 2: w staging reuses [0, 128*NOUT) after a barrier (W3T untouched).
// ---------------------------------------------------------------------------
template <int NT3, bool HAS_X1>
__global__ __launch_bounds__(256)
void edge_fused_kernel(const _Float16* __restrict__ embh,  // [E][8]
                       const float* __restrict__ Y1,       // [E][3]
                       const int* __restrict__ ei,
                       const int* __restrict__ ej,
                       const float* __restrict__ fc0,      // [8][64]
                       const float* __restrict__ fc1,      // [64][64]
                       const float* __restrict__ fc2,      // [64][fc2cols]
                       int fc2cols,
                       const float* __restrict__ h0,       // [N][32]
                       const float* __restrict__ h1,       // [N][32][3] (opt)
                       float* __restrict__ a0,             // [N][A0C]
                       float* __restrict__ a1,             // [N][A1C][3]
                       int nE) {
  constexpr int NOUT = NT3 * 16;
  constexpr int A0C  = HAS_X1 ? 64 : 32;
  constexpr int A1C  = HAS_X1 ? 64 : 32;

  __shared__ _Float16 smem[26624];  // 52 KB
  _Float16* emb_s = smem;
  _Float16* W1T   = smem + 4096;
  _Float16* W2T   = smem + 6144;
  _Float16* hstg  = smem + 10240;
  _Float16* W3T   = smem + 18432;

  const int tid   = threadIdx.x;
  const int lane  = tid & 31;
  const int wave  = tid >> 5;
  const int eBase = blockIdx.x * 128;

  // --- stage inputs + f16-transposed weights ---
  for (int i = tid; i < 128 * 32; i += 256) {
    int row = i >> 5, k = i & 31;
    int e = eBase + row;
    float v = (k < NBASIS && e < nE) ? (float)embh[e * NBASIS + k] : 0.0f;
    emb_s[i] = (_Float16)v;  // K padded 8 -> 32 with zeros
  }
  for (int i = tid; i < 64 * 32; i += 256) {
    int o = i >> 5, k = i & 31;
    W1T[i] = (_Float16)((k < NBASIS) ? fc0[k * 64 + o] : 0.0f);
  }
  for (int i = tid; i < 64 * 64; i += 256) {
    int o = i >> 6, k = i & 63;
    W2T[i] = (_Float16)fc1[k * 64 + o];
  }
  for (int i = tid; i < NOUT * 64; i += 256) {
    int o = i >> 6, k = i & 63;
    W3T[i] = (_Float16)fc2[k * fc2cols + o];
  }
  __syncthreads();

  const int ncol  = lane & 15;        // D layout: lane -> column N
  const int mbase = (lane >> 4) * 8;  // D layout: vgpr r -> row mbase+r
  _Float16* hw = hstg + wave * (16 * 64);

  // --- layer 1: [16x32(pad)] x [32x64] ---
  {
    v16h af = frag_ld(emb_s + wave * (16 * 32), 32, 0, lane);
#pragma unroll
    for (int nt = 0; nt < 4; ++nt) {
      v16h bf = frag_ld(W1T + nt * (16 * 32), 32, 0, lane);
      v8f z = {};
      z = wmma_f16(af, bf, z);
#pragma unroll
      for (int r = 0; r < 8; ++r)
        hw[(mbase + r) * 64 + nt * 16 + ncol] = (_Float16)siluf(z[r]);
    }
  }
  // --- layer 2: [16x64] x [64x64] ---
  {
    v16h af0 = frag_ld(hw, 64, 0, lane);
    v16h af1 = frag_ld(hw, 64, 32, lane);
#pragma unroll
    for (int nt = 0; nt < 4; ++nt) {
      v16h b0 = frag_ld(W2T + nt * (16 * 64), 64, 0, lane);
      v16h b1 = frag_ld(W2T + nt * (16 * 64), 64, 32, lane);
      v8f z = {};
      z = wmma_f16(af0, b0, z);
      z = wmma_f16(af1, b1, z);
#pragma unroll
      for (int r = 0; r < 8; ++r)
        hw[(mbase + r) * 64 + nt * 16 + ncol] = (_Float16)siluf(z[r]);
    }
  }
  // --- layer 3: [16x64] x [64xNOUT], result -> LDS w staging ---
  v16h a30 = frag_ld(hw, 64, 0, lane);
  v16h a31 = frag_ld(hw, 64, 32, lane);
  __syncthreads();  // all A-frags in regs: emb/W1T/W2T/hstg regions reusable
  _Float16* wrow = smem + wave * (16 * NOUT);
#pragma unroll
  for (int nt = 0; nt < NT3; ++nt) {
    v16h b0 = frag_ld(W3T + nt * (16 * 64), 64, 0, lane);
    v16h b1 = frag_ld(W3T + nt * (16 * 64), 64, 32, lane);
    v8f z = {};
    z = wmma_f16(a30, b0, z);
    z = wmma_f16(a31, b1, z);
#pragma unroll
    for (int r = 0; r < 8; ++r)
      wrow[(mbase + r) * NOUT + nt * 16 + ncol] = (_Float16)z[r];
  }
  __syncthreads();

  // --- phase 2: messages + atomic segment-sum (L2-resident accumulators) ---
  const float Y0C = 0.28209479177387814f;  // 1/(2*sqrt(pi))
  const float IS3 = 0.57735026918962576f;  // 1/sqrt(3)
  for (int i = tid; i < 128 * MULC; i += 256) {
    int el = i >> 5, u = i & 31;
    int e = eBase + el;
    if (e >= nE) continue;
    const _Float16* wv = smem + el * NOUT;
    int src = ei[e], dst = ej[e];
    float y0 = Y1[e * 3 + 0], y1v = Y1[e * 3 + 1], y2 = Y1[e * 3 + 2];
    float xi0 = h0[src * MULC + u];
    float w0 = (float)wv[u];
    float w1 = (float)wv[MULC + u];
    atomicAdd(&a0[dst * A0C + u], w0 * xi0 * Y0C);
    float f1 = w1 * xi0;
    atomicAdd(&a1[(dst * A1C + u) * 3 + 0], f1 * y0);
    atomicAdd(&a1[(dst * A1C + u) * 3 + 1], f1 * y1v);
    atomicAdd(&a1[(dst * A1C + u) * 3 + 2], f1 * y2);
    if (HAS_X1) {
      float w2 = (float)wv[2 * MULC + u];
      float w3 = (float)wv[3 * MULC + u];
      float xa = h1[(src * MULC + u) * 3 + 0];
      float xb = h1[(src * MULC + u) * 3 + 1];
      float xc = h1[(src * MULC + u) * 3 + 2];
      atomicAdd(&a1[(dst * A1C + MULC + u) * 3 + 0], w2 * xa * Y0C);
      atomicAdd(&a1[(dst * A1C + MULC + u) * 3 + 1], w2 * xb * Y0C);
      atomicAdd(&a1[(dst * A1C + MULC + u) * 3 + 2], w2 * xc * Y0C);
      float dv = xa * y0 + xb * y1v + xc * y2;
      atomicAdd(&a0[dst * A0C + MULC + u], w3 * dv * IS3);
    }
  }
}

// ---------------------------------------------------------------------------
// Node update: z0 = a0 @ W2_0 + sc0 ; z1 = einsum(a1, W2_1) + sc1 ;
// x0 (+)= silu(z0[:32]) ; x1 (+)= silu(z0[32:])[:,None] * z1
// ---------------------------------------------------------------------------
template <bool FIRST>
__global__ __launch_bounds__(64)
void node_update_kernel(const float* __restrict__ a0,
                        const float* __restrict__ a1,
                        const float* __restrict__ sc0,
                        const float* __restrict__ sc1,
                        const float* __restrict__ W2_0,
                        const float* __restrict__ W2_1,
                        float* __restrict__ x0,
                        float* __restrict__ x1,
                        int n) {
  constexpr int A0C = FIRST ? 32 : 64;
  constexpr int A1C = FIRST ? 32 : 64;
  int node = blockIdx.x;
  if (node >= n) return;
  int tid = threadIdx.x;
  __shared__ float a0s[64];
  __shared__ float a1s[192];
  __shared__ float z0s[64];
  for (int i = tid; i < A0C; i += 64) a0s[i] = a0[node * A0C + i];
  for (int i = tid; i < A1C * 3; i += 64) a1s[i] = a1[node * A1C * 3 + i];
  __syncthreads();
  float acc = sc0[node * 64 + tid];
#pragma unroll 4
  for (int u = 0; u < A0C; ++u) acc += a0s[u] * W2_0[u * 64 + tid];
  z0s[tid] = acc;
  float z1m0 = 0.0f, z1m1 = 0.0f, z1m2 = 0.0f;
  if (tid < 32) {
#pragma unroll 4
    for (int u = 0; u < A1C; ++u) {
      float wv = W2_1[u * 32 + tid];
      z1m0 += a1s[u * 3 + 0] * wv;
      z1m1 += a1s[u * 3 + 1] * wv;
      z1m2 += a1s[u * 3 + 2] * wv;
    }
    if (!FIRST) {
      z1m0 += sc1[(node * 32 + tid) * 3 + 0];
      z1m1 += sc1[(node * 32 + tid) * 3 + 1];
      z1m2 += sc1[(node * 32 + tid) * 3 + 2];
    }
  }
  __syncthreads();
  if (tid < 32) {
    float s = siluf(z0s[tid]);
    float g = siluf(z0s[32 + tid]);
    int b0 = node * 32 + tid;
    if (FIRST) {
      x0[b0] = s;
      x1[b0 * 3 + 0] = g * z1m0;
      x1[b0 * 3 + 1] = g * z1m1;
      x1[b0 * 3 + 2] = g * z1m2;
    } else {
      x0[b0] += s;
      x1[b0 * 3 + 0] += g * z1m0;
      x1[b0 * 3 + 1] += g * z1m1;
      x1[b0 * 3 + 2] += g * z1m2;
    }
  }
}

// ---------------------------------------------------------------------------
// Readout: e[n] = (x0 @ Wr1) @ Wr2 + atom_energies[t]; atomic segment sum.
// ---------------------------------------------------------------------------
__global__ __launch_bounds__(256)
void readout_kernel(const float* __restrict__ x0,
                    const float* __restrict__ Wr1,
                    const float* __restrict__ Wr2,
                    const float* __restrict__ atomE,
                    const int* __restrict__ types,
                    const int* __restrict__ batch,
                    float* __restrict__ out, int n) {
  int node = blockIdx.x * 256 + threadIdx.x;
  if (node >= n) return;
  float acc = atomE[types[node]];
#pragma unroll
  for (int k = 0; k < 16; ++k) {
    float t = 0.0f;
#pragma unroll
    for (int u = 0; u < 32; ++u) t += x0[node * 32 + u] * Wr1[u * 16 + k];
    acc += t * Wr2[k];
  }
  atomicAdd(&out[batch[node]], acc);
}

// ---------------------------------------------------------------------------
// Host launch
// ---------------------------------------------------------------------------
extern "C" void kernel_launch(void* const* d_in, const int* in_sizes, int n_in,
                              void* d_out, int out_size, void* d_ws, size_t ws_size,
                              hipStream_t stream) {
  (void)n_in; (void)ws_size;
  // Assumed flatten order: recursive insertion order of setup_inputs() dict.
  enum {
    I_POS = 0, I_EMB, I_BPI,
    I_B0_W10, I_B0_FC0, I_B0_FC1, I_B0_FC2, I_B0_W20, I_B0_W21, I_B0_WSC0,
    I_B1_W10, I_B1_W11, I_B1_FC0, I_B1_FC1, I_B1_FC2, I_B1_W20, I_B1_W21,
    I_B1_WSC0, I_B1_WSC1,
    I_B2_W10, I_B2_W11, I_B2_FC0, I_B2_FC1, I_B2_FC2, I_B2_W20, I_B2_W21,
    I_B2_WSC0, I_B2_WSC1,
    I_WR1, I_WR2, I_AE, I_TYPES, I_EDGE, I_BATCH
  };
  const int E = in_sizes[I_EDGE] / 2;
  const int N = in_sizes[I_TYPES];

  const float* pos   = (const float*)d_in[I_POS];
  const float* embT  = (const float*)d_in[I_EMB];
  const float* bpi   = (const float*)d_in[I_BPI];
  const int*   types = (const int*)d_in[I_TYPES];
  const int*   edge  = (const int*)d_in[I_EDGE];
  const int*   batch = (const int*)d_in[I_BATCH];
  const int*   ei    = edge;
  const int*   ej    = edge + E;

  // workspace bump allocator (256B aligned)
  uintptr_t base = (uintptr_t)d_ws;
  size_t off = 0;
  auto alloc = [&](size_t bytes) -> void* {
    void* p = (void*)(base + off);
    off = (off + bytes + 255) & ~(size_t)255;
    return p;
  };
  float*    Y1   = (float*)alloc((size_t)E * 3 * sizeof(float));
  _Float16* embh = (_Float16*)alloc((size_t)E * NBASIS * sizeof(_Float16));
  float*    x0   = (float*)alloc((size_t)N * 32 * sizeof(float));
  float*    x1   = (float*)alloc((size_t)N * 96 * sizeof(float));
  float*    h0   = (float*)alloc((size_t)N * 32 * sizeof(float));
  float*    h1   = (float*)alloc((size_t)N * 96 * sizeof(float));
  float*    sc0  = (float*)alloc((size_t)N * 64 * sizeof(float));
  float*    sc1  = (float*)alloc((size_t)N * 96 * sizeof(float));
  float*    a0   = (float*)alloc((size_t)N * 64 * sizeof(float));
  float*    a1   = (float*)alloc((size_t)N * 192 * sizeof(float));

  float* out = (float*)d_out;
  fill_zero_kernel<<<(out_size + 255) / 256, 256, 0, stream>>>(out, out_size);

  edge_geom_kernel<<<(E + 255) / 256, 256, 0, stream>>>(pos, ei, ej, bpi, Y1, embh, E);
  node_embed_kernel<<<(N * 32 + 255) / 256, 256, 0, stream>>>(embT, types, x0, N * 32);

  const int eg = (E + 127) / 128;

  // ---- block 0 ----
  node_pre_kernel<false><<<N, 64, 0, stream>>>(
      x0, nullptr, types,
      (const float*)d_in[I_B0_W10], nullptr,
      (const float*)d_in[I_B0_WSC0], nullptr,
      h0, nullptr, sc0, nullptr, N);
  fill_zero_kernel<<<(N * 32 + 255) / 256, 256, 0, stream>>>(a0, N * 32);
  fill_zero_kernel<<<(N * 96 + 255) / 256, 256, 0, stream>>>(a1, N * 96);
  edge_fused_kernel<4, false><<<eg, 256, 0, stream>>>(
      embh, Y1, ei, ej,
      (const float*)d_in[I_B0_FC0], (const float*)d_in[I_B0_FC1],
      (const float*)d_in[I_B0_FC2], 64,
      h0, nullptr, a0, a1, E);
  node_update_kernel<true><<<N, 64, 0, stream>>>(
      a0, a1, sc0, nullptr,
      (const float*)d_in[I_B0_W20], (const float*)d_in[I_B0_W21],
      x0, x1, N);

  // ---- blocks 1 and 2 ----
  const int blkIdx[2][9] = {
      {I_B1_W10, I_B1_W11, I_B1_FC0, I_B1_FC1, I_B1_FC2, I_B1_W20, I_B1_W21,
       I_B1_WSC0, I_B1_WSC1},
      {I_B2_W10, I_B2_W11, I_B2_FC0, I_B2_FC1, I_B2_FC2, I_B2_W20, I_B2_W21,
       I_B2_WSC0, I_B2_WSC1}};
  for (int b = 0; b < 2; ++b) {
    const float* W10  = (const float*)d_in[blkIdx[b][0]];
    const float* W11  = (const float*)d_in[blkIdx[b][1]];
    const float* FC0  = (const float*)d_in[blkIdx[b][2]];
    const float* FC1  = (const float*)d_in[blkIdx[b][3]];
    const float* FC2  = (const float*)d_in[blkIdx[b][4]];
    const float* W20  = (const float*)d_in[blkIdx[b][5]];
    const float* W21  = (const float*)d_in[blkIdx[b][6]];
    const float* WSC0 = (const float*)d_in[blkIdx[b][7]];
    const float* WSC1 = (const float*)d_in[blkIdx[b][8]];
    node_pre_kernel<true><<<N, 64, 0, stream>>>(
        x0, x1, types, W10, W11, WSC0, WSC1, h0, h1, sc0, sc1, N);
    fill_zero_kernel<<<(N * 64 + 255) / 256, 256, 0, stream>>>(a0, N * 64);
    fill_zero_kernel<<<(N * 192 + 255) / 256, 256, 0, stream>>>(a1, N * 192);
    // MLP output is 6*MUL wide but only first 4*MUL columns are consumed.
    edge_fused_kernel<8, true><<<eg, 256, 0, stream>>>(
        embh, Y1, ei, ej, FC0, FC1, FC2, 192, h0, h1, a0, a1, E);
    node_update_kernel<false><<<N, 64, 0, stream>>>(
        a0, a1, sc0, sc1, W20, W21, x0, x1, N);
  }

  // ---- readout ----
  readout_kernel<<<(N + 255) / 256, 256, 0, stream>>>(
      x0, (const float*)d_in[I_WR1], (const float*)d_in[I_WR2],
      (const float*)d_in[I_AE], types, batch, out, N);
}
